// PredictionNet_86500641342132
// MI455X (gfx1250) — matrix operand order repaired and checked by reading, not compiled, not run on hardware
//
#include <hip/hip_runtime.h>
#include <math.h>

typedef __attribute__((ext_vector_type(16))) _Float16 v16h;
typedef __attribute__((ext_vector_type(8)))  _Float16 v8h;
typedef __attribute__((ext_vector_type(4)))  _Float16 v4h;
typedef __attribute__((ext_vector_type(8)))  float    v8f;
typedef __attribute__((ext_vector_type(4)))  float    v4f;

#define EPS 1e-6f

__global__ void __launch_bounds__(256, 1)
clifford_prediction_kernel(const float* __restrict__ state,     // [B,32,8]
                           const float* __restrict__ ln_gamma,  // [32,4]
                           const float* __restrict__ W_lin,     // [16,32,4]
                           const float* __restrict__ b_lin,     // [16]
                           const float* __restrict__ act_a,     // [16,4]
                           const float* __restrict__ act_b,     // [16]
                           const float* __restrict__ value_w,   // [16,1]
                           const float* __restrict__ value_b,   // [1]
                           const float* __restrict__ policy_w,  // [48,5]
                           const float* __restrict__ policy_b,  // [5]
                           float* __restrict__ out_policy,      // [B,5]
                           float* __restrict__ out_value,       // [B]
                           int B)
{
  // Per-wave staging for the fused head matmul operand (A-layout [M=16][K=64] f16).
  __shared__ _Float16 lds_act[8][16][64];   // 8 waves x 2 KB

  const int lane    = threadIdx.x & 31;
  const int wib     = threadIdx.x >> 5;
  const int wave_id = blockIdx.x * (blockDim.x >> 5) + wib;
  const int nwaves  = gridDim.x * (blockDim.x >> 5);

  const int r = lane & 15;                  // A-fragment row M owned by this lane
  const int h = lane >> 4;                  // lane half
  const int o = lane & 15;                  // B/C-fragment column N

  // ================= one-time per-wave setup =================
  // B operand for CliffordLinear: W_lin with ln_gamma folded, one fragment per grade.
  v16h bfrag[4];
  #pragma unroll
  for (int e = 0; e < 16; ++e) {
    const int c = (e < 8) ? (8 * h + e) : (8 + 8 * h + e);
    const v4f w = *(const v4f*)(W_lin + ((o << 5) + c) * 4);   // W_lin[o,c,0..3]
    const v4f g = *(const v4f*)(ln_gamma + c * 4);             // gamma[c,0..3]
    bfrag[0][e] = (_Float16)(w.x * g.x);
    bfrag[1][e] = (_Float16)(w.y * g.y);
    bfrag[2][e] = (_Float16)(w.z * g.z);
    bfrag[3][e] = (_Float16)(w.w * g.w);
  }

  // B operand for fused heads: W2[K=64, N=16], K = 4*o2 + k (branchless build).
  //   k=0..2 : grade-1 blades -> policy_w[(o2*3+k)*5 + a], columns a=0..4
  //   k=3    : scalar blade   -> value_w[o2], column a=5
  const float mp = (o < 5)  ? 1.0f : 0.0f;
  const float mv = (o == 5) ? 1.0f : 0.0f;
  const int   oc = (o < 5)  ? o : 4;        // clamped column for safe loads
  v16h hfrag[2];
  #pragma unroll
  for (int w = 0; w < 2; ++w) {
    #pragma unroll
    for (int e = 0; e < 16; ++e) {
      const int K  = ((e < 8) ? (8 * h + e) : (8 + 8 * h + e)) + 32 * w;   // compile-time constant after unroll
      const int o2 = K >> 2, k = K & 3;
      const float v = (k < 3) ? (policy_w[(o2 * 3 + k) * 5 + oc] * mp)
                              : (value_w[o2] * mv);
      hfrag[w][e] = (_Float16)v;
    }
  }

  // Per-lane constants.
  const float bl = b_lin[o];
  const v4f  aa = *(const v4f*)(act_a + o * 4);
  const float ab = act_b[o];
  const float ob = (o < 5) ? policy_b[o] : ((o == 5) ? value_b[0] : 0.0f);
  const v8f z = {};

  // ================= persistent loop over 16-row tiles =================
  const int ntiles = (B + 15) >> 4;
  for (int tile = wave_id; tile < ntiles; tile += nwaves) {
    const int row0 = tile << 4;

    // Prefetch this wave's next tile (16 KB = 128 lines), wave-uniform guard.
    const int nt = tile + nwaves;
    if (nt < ntiles) {
      const char* np = (const char*)(state + (size_t)nt * 4096);
      #pragma unroll
      for (int i = 0; i < 4; ++i)
        __builtin_prefetch(np + (size_t)(lane + 32 * i) * 128, 0, 3);
    }

    // A operand mapping (16x32 f16): lane holds M = lane%16; element e <-> channel
    //   c = e<8 ? 8h+e : 16+8h+(e-8). Identical (lane,e) mapping for every blade.
    const int brow   = row0 + r;
    const int brow_c = brow < B ? brow : B - 1;   // clamp keeps EXEC full for WMMA
    const float* sp = state + (size_t)brow_c * 256;

    v8f acc[8];

    // ---- pass 1: blades 0..3 (scalar + grade-1), needs only x[0..3] and n0,n1.
    {
      v16h af[4];
      #pragma unroll
      for (int e = 0; e < 16; ++e) {
        const int c = (e < 8) ? (8 * h + e) : (8 + 8 * h + e);
        const v4f x = *(const v4f*)(sp + c * 8);
        float s0 = fmaf(x.x, x.x, EPS);
        float s1 = fmaf(x.y, x.y, EPS);
        s1 = fmaf(x.z, x.z, s1);
        s1 = fmaf(x.w, x.w, s1);
        const float i0 = rsqrtf(s0);
        const float i1 = rsqrtf(s1);
        af[0][e] = (_Float16)(x.x * i0);
        af[1][e] = (_Float16)(x.y * i1);
        af[2][e] = (_Float16)(x.z * i1);
        af[3][e] = (_Float16)(x.w * i1);
      }
      acc[0] = __builtin_amdgcn_wmma_f32_16x16x32_f16(false, af[0], false, bfrag[0], (short)0, z, false, false);
      acc[1] = __builtin_amdgcn_wmma_f32_16x16x32_f16(false, af[1], false, bfrag[1], (short)0, z, false, false);
      acc[2] = __builtin_amdgcn_wmma_f32_16x16x32_f16(false, af[2], false, bfrag[1], (short)0, z, false, false);
      acc[3] = __builtin_amdgcn_wmma_f32_16x16x32_f16(false, af[3], false, bfrag[1], (short)0, z, false, false);
    }

    // ---- pass 2: blades 4..7 (grade-2 + pseudoscalar), needs only x[4..7] and n2,n3.
    {
      v16h af[4];
      #pragma unroll
      for (int e = 0; e < 16; ++e) {
        const int c = (e < 8) ? (8 * h + e) : (8 + 8 * h + e);
        const v4f x = *(const v4f*)(sp + c * 8 + 4);
        float s2 = fmaf(x.x, x.x, EPS);
        s2 = fmaf(x.y, x.y, s2);
        s2 = fmaf(x.z, x.z, s2);
        float s3 = fmaf(x.w, x.w, EPS);
        const float i2 = rsqrtf(s2);
        const float i3 = rsqrtf(s3);
        af[0][e] = (_Float16)(x.x * i2);
        af[1][e] = (_Float16)(x.y * i2);
        af[2][e] = (_Float16)(x.z * i2);
        af[3][e] = (_Float16)(x.w * i3);
      }
      acc[4] = __builtin_amdgcn_wmma_f32_16x16x32_f16(false, af[0], false, bfrag[2], (short)0, z, false, false);
      acc[5] = __builtin_amdgcn_wmma_f32_16x16x32_f16(false, af[1], false, bfrag[2], (short)0, z, false, false);
      acc[6] = __builtin_amdgcn_wmma_f32_16x16x32_f16(false, af[2], false, bfrag[2], (short)0, z, false, false);
      acc[7] = __builtin_amdgcn_wmma_f32_16x16x32_f16(false, af[3], false, bfrag[3], (short)0, z, false, false);
    }

    // ---- GeometricGELU gate; stage gated activations in LDS head-A layout.
    // C/D layout: lane holds N=o, rows M = 8h + j. One ds_store_b64 per row (K=4o..4o+3).
    #pragma unroll
    for (int j = 0; j < 8; ++j) {
      const int m = 8 * h + j;
      const float y0 = acc[0][j] + bl;     // bias on scalar blade
      const float y1 = acc[1][j], y2 = acc[2][j], y3 = acc[3][j];
      const float y4 = acc[4][j], y5 = acc[5][j], y6 = acc[6][j], y7 = acc[7][j];
      float s0 = fmaf(y0, y0, EPS);
      float s1 = fmaf(y1, y1, EPS); s1 = fmaf(y2, y2, s1); s1 = fmaf(y3, y3, s1);
      float s2 = fmaf(y4, y4, EPS); s2 = fmaf(y5, y5, s2); s2 = fmaf(y6, y6, s2);
      float s3 = fmaf(y7, y7, EPS);
      float q = ab;
      q = fmaf(sqrtf(s0), aa.x, q);
      q = fmaf(sqrtf(s1), aa.y, q);
      q = fmaf(sqrtf(s2), aa.z, q);
      q = fmaf(sqrtf(s3), aa.w, q);
      const float gate = 0.5f * (1.0f + erff(q * 0.70710678118654752f));
      v4h pk;
      pk[0] = (_Float16)(y1 * gate);       // k=0..2: grade-1 blades
      pk[1] = (_Float16)(y2 * gate);
      pk[2] = (_Float16)(y3 * gate);
      pk[3] = (_Float16)(y0 * gate);       // k=3: scalar blade (value path)
      *(v4h*)&lds_act[wib][m][4 * o] = pk;
    }

    // ---- Fused heads: OUT[16,16] = A2[16,64] x W2[64,16], two chained WMMAs.
    v8h lo0 = *(const v8h*)&lds_act[wib][r][8 * h];
    v8h hi0 = *(const v8h*)&lds_act[wib][r][16 + 8 * h];
    v8h lo1 = *(const v8h*)&lds_act[wib][r][32 + 8 * h];
    v8h hi1 = *(const v8h*)&lds_act[wib][r][48 + 8 * h];
    v16h ha0 = __builtin_shufflevector(lo0, hi0, 0,1,2,3,4,5,6,7,8,9,10,11,12,13,14,15);
    v16h ha1 = __builtin_shufflevector(lo1, hi1, 0,1,2,3,4,5,6,7,8,9,10,11,12,13,14,15);
    v8f hout;
    hout = __builtin_amdgcn_wmma_f32_16x16x32_f16(false, ha0, false, hfrag[0], (short)0, z,    false, false);
    hout = __builtin_amdgcn_wmma_f32_16x16x32_f16(false, ha1, false, hfrag[1], (short)0, hout, false, false);

    // ---- Outputs: lane column a = o (a<5: policy, a==5: value); rows m = 8h+j.
    if (o < 6) {
      #pragma unroll
      for (int j = 0; j < 8; ++j) {
        const int grow = row0 + 8 * h + j;
        if (grow < B) {
          const float val = hout[j] + ob;
          if (o < 5) out_policy[(size_t)grow * 5 + o] = val;
          else       out_value[grow] = val;
        }
      }
    }
  }
}

extern "C" void kernel_launch(void* const* d_in, const int* in_sizes, int n_in,
                              void* d_out, int out_size, void* d_ws, size_t ws_size,
                              hipStream_t stream) {
  const float* state    = (const float*)d_in[0];
  const float* ln_gamma = (const float*)d_in[1];
  const float* W_lin    = (const float*)d_in[2];
  const float* b_lin    = (const float*)d_in[3];
  const float* act_a    = (const float*)d_in[4];
  const float* act_b    = (const float*)d_in[5];
  const float* value_w  = (const float*)d_in[6];
  const float* value_b  = (const float*)d_in[7];
  const float* policy_w = (const float*)d_in[8];
  const float* policy_b = (const float*)d_in[9];

  const int B = in_sizes[0] / 256;          // state is [B, 32, 8]
  float* out        = (float*)d_out;
  float* out_policy = out;                  // [B,5] first in return order
  float* out_value  = out + (size_t)B * 5;  // then [B,1]

  // Persistent waves: ~4 tiles per wave, capped at 2048 waves (256 blocks x 8 waves).
  const long waves_needed = ((long)B + 15) / 16;
  long target_waves = (waves_needed + 3) / 4;
  if (target_waves > 2048) target_waves = 2048;
  if (target_waves < 1)    target_waves = 1;
  const int blocks = (int)((target_waves + 7) / 8);

  clifford_prediction_kernel<<<blocks, 256, 0, stream>>>(
      state, ln_gamma, W_lin, b_lin, act_a, act_b,
      value_w, value_b, policy_w, policy_b,
      out_policy, out_value, B);
}
